// Decoder_51479478010003
// MI455X (gfx1250) — matrix operand order repaired and checked by reading, not compiled, not run on hardware
//
#include <hip/hip_runtime.h>
#include <hip/hip_bf16.h>

typedef __attribute__((ext_vector_type(16))) _Float16 v16h;
typedef __attribute__((ext_vector_type(8)))  _Float16 v8h;
typedef __attribute__((ext_vector_type(4)))  _Float16 v4h;
typedef __attribute__((ext_vector_type(8)))  float    v8f;
typedef __attribute__((ext_vector_type(4)))  float    v4f;

// Problem constants
static constexpr int Bc   = 16;
static constexpr int Nc   = 32;
static constexpr int Ec   = 256;
static constexpr int Dc   = 256;
static constexpr int Hc   = 8;
static constexpr int HWc  = 784;
static constexpr int Lc   = 4;
static constexpr int DHc  = 32;
static constexpr int FFc  = 1024;
static constexpr int NEc  = 288;            // N + E tokens
static constexpr int SR   = Bc * NEc;       // 4608 token rows
static constexpr int IR   = Bc * HWc;       // 12544 image rows
static constexpr int BNc  = Bc * Nc;        // 512
static constexpr int BEc  = Bc * Ec;        // 4096
static constexpr float SCALEc = 0.17677669529663687f; // 1/sqrt(32)

__device__ __forceinline__ float geluf(float x) {
  return 0.5f * x * (1.0f + erff(x * 0.70710678118654752f));
}

__device__ __forceinline__ v8f wmma16(v16h a, v16h b, v8f c) {
  return __builtin_amdgcn_wmma_f32_16x16x32_f16(false, a, false, b, (short)0, c, false, false);
}

__device__ __forceinline__ v16h cat8(v8h lo, v8h hi) {
  return __builtin_shufflevector(lo, hi, 0, 1, 2, 3, 4, 5, 6, 7, 8, 9, 10, 11, 12, 13, 14, 15);
}

__device__ __forceinline__ void atomicMaxF(float* addr, float val) {
  unsigned* ua = (unsigned*)addr;
  unsigned old = *ua;
  while (true) {
    float f = __uint_as_float(old);
    if (f >= val) break;
    unsigned assumed = old;
    old = atomicCAS(ua, assumed, __float_as_uint(val));
    if (old == assumed) break;
  }
}

// ---------------------------------------------------------------------------
// WMMA GEMM: C[M,N] = epilogue(A[M,K] @ W[K,N] + bias)
// All shapes in this model are tile-aligned: M%32==0, N%128==0, K%32==0 ->
// no bounds checks, b128 global loads, b128 LDS fragment reads.
// Block = 128 threads (4 waves); block tile 32x128.
// Wave w computes a 16x64 slab: rows (w&1)*16, cols (w>>1)*64 -> one
// A-fragment reused across 4 WMMAs per K-step (compute amplification).
// mode 0: store   mode 1: gelu   mode 2: C = res + ls[col]*val
// ---------------------------------------------------------------------------
__global__ __launch_bounds__(128) void k_gemm(
    const float* __restrict__ A, int lda,
    const float* __restrict__ W, int ldw,
    const float* __restrict__ bias,
    float* __restrict__ C, int ldc,
    int M, int Nn, int K,
    int mode, const float* __restrict__ res, const float* __restrict__ ls)
{
  __shared__ alignas(16) _Float16 As[32 * 32];    // row-major [row][k]
  __shared__ alignas(16) _Float16 Bsw[128 * 32];  // transposed [col][k]
  const int tid   = threadIdx.x;
  const int wave  = tid >> 5;
  const int lane  = tid & 31;
  const int mrow  = lane & 15;
  const int g     = lane >> 4;
  const int rbase = (wave & 1) * 16;
  const int cbase = (wave >> 1) * 64;
  const int m0    = blockIdx.x * 32;
  const int n0    = blockIdx.y * 128;

  v8f acc0 = {}, acc1 = {}, acc2 = {}, acc3 = {};
  for (int k0 = 0; k0 < K; k0 += 32) {
    if (k0 + 32 < K)
      __builtin_prefetch(&W[(size_t)(k0 + 32) * ldw + n0], 0, 0);
    // A tile 32x32: 1024 halves, two b128 loads per thread
#pragma unroll
    for (int i = 0; i < 2; ++i) {
      int idx = tid + i * 128;        // 0..255
      int r = idx >> 3, c = (idx & 7) * 4;
      v4f fv = *(const v4f*)&A[(size_t)(m0 + r) * lda + k0 + c];
      *(v4h*)&As[r * 32 + c] = __builtin_convertvector(fv, v4h);
    }
    // B tile 32x128: coalesced b128 loads, transpose-scatter into Bsw[col][k]
#pragma unroll
    for (int i = 0; i < 8; ++i) {
      int idx = tid + i * 128;        // 0..1023
      int r = idx >> 5;               // k row 0..31
      int c4 = (idx & 31) * 4;        // col group 0..124
      v4f fv = *(const v4f*)&W[(size_t)(k0 + r) * ldw + n0 + c4];
#pragma unroll
      for (int j = 0; j < 4; ++j) Bsw[(c4 + j) * 32 + r] = (_Float16)fv[j];
    }
    __syncthreads();
    // A frag: two contiguous 8-half runs per ISA 16-bit A(16x32) layout
    v16h a = cat8(*(const v8h*)&As[(rbase + mrow) * 32 + 8 * g],
                  *(const v8h*)&As[(rbase + mrow) * 32 + 16 + 8 * g]);
    // 4 B frags / 4 WMMAs reusing the same A fragment
    int bc = cbase + mrow;
    v16h b0 = cat8(*(const v8h*)&Bsw[(bc +  0) * 32 + 8 * g],
                   *(const v8h*)&Bsw[(bc +  0) * 32 + 16 + 8 * g]);
    v16h b1 = cat8(*(const v8h*)&Bsw[(bc + 16) * 32 + 8 * g],
                   *(const v8h*)&Bsw[(bc + 16) * 32 + 16 + 8 * g]);
    v16h b2 = cat8(*(const v8h*)&Bsw[(bc + 32) * 32 + 8 * g],
                   *(const v8h*)&Bsw[(bc + 32) * 32 + 16 + 8 * g]);
    v16h b3 = cat8(*(const v8h*)&Bsw[(bc + 48) * 32 + 8 * g],
                   *(const v8h*)&Bsw[(bc + 48) * 32 + 16 + 8 * g]);
    acc0 = wmma16(a, b0, acc0);
    acc1 = wmma16(a, b1, acc1);
    acc2 = wmma16(a, b2, acc2);
    acc3 = wmma16(a, b3, acc3);
    __syncthreads();
  }
  // C layout: VGPR r -> row r (lanes 0-15) / r+8 (lanes 16-31), col = lane&15
#pragma unroll
  for (int r = 0; r < 8; ++r) {
    int row = m0 + rbase + r + 8 * g;
    float vv[4] = {acc0[r], acc1[r], acc2[r], acc3[r]};
#pragma unroll
    for (int s = 0; s < 4; ++s) {
      int col = n0 + cbase + s * 16 + mrow;
      float v = vv[s] + bias[col];
      if (mode == 1) v = geluf(v);
      float outv = (mode == 2) ? (res[(size_t)row * ldc + col] + ls[col] * v) : v;
      C[(size_t)row * ldc + col] = outv;
    }
  }
}

// ---------------------------------------------------------------------------
// Fused cross-attention with online softmax. One wave per (q-tile, head, b).
// Per 32-key step: 2x WMMA scores (q @ kT), softmax update, 2x WMMA ctx (P@V).
// K staged row-major (kT frag is contiguous), V staged transposed in LDS.
// ---------------------------------------------------------------------------
__global__ __launch_bounds__(32) void k_attn(
    const float* __restrict__ qb, const float* __restrict__ kb,
    const float* __restrict__ vb, const float* __restrict__ mask,
    float* __restrict__ ctx)
{
  const int qt = blockIdx.x, h = blockIdx.y, b = blockIdx.z;
  const int lane = threadIdx.x & 31;
  const int m = lane & 15;
  const int g = lane >> 4;
  const int q0 = qt * 16;

  __shared__ alignas(16) _Float16 kl[32 * 32];   // [key][dh]
  __shared__ alignas(16) _Float16 vlT[32 * 32];  // [dh][key]
  __shared__ alignas(16) _Float16 pl[16 * 32];   // [q][key]

  // q A-fragment: halves are runs qrow[8g..8g+7] and qrow[16+8g..16+8g+7]
  v16h aq;
  {
    const float* qrow = qb + (size_t)(b * NEc + q0 + m) * Dc + h * DHc;
    v4f f0 = *(const v4f*)&qrow[8 * g];
    v4f f1 = *(const v4f*)&qrow[8 * g + 4];
    v4f f2 = *(const v4f*)&qrow[16 + 8 * g];
    v4f f3 = *(const v4f*)&qrow[16 + 8 * g + 4];
    v8h lo = __builtin_shufflevector(__builtin_convertvector(f0, v4h),
                                     __builtin_convertvector(f1, v4h),
                                     0, 1, 2, 3, 4, 5, 6, 7);
    v8h hi = __builtin_shufflevector(__builtin_convertvector(f2, v4h),
                                     __builtin_convertvector(f3, v4h),
                                     0, 1, 2, 3, 4, 5, 6, 7);
    aq = cat8(lo, hi);
  }

  float mrun[8], lrun[8];
  v8f acc0 = {}, acc1 = {};
#pragma unroll
  for (int r = 0; r < 8; ++r) { mrun[r] = -1e30f; lrun[r] = 0.f; }

  const float* mbase = mask + (size_t)(b * Hc + h) * NEc * HWc;
  const v4f zero4 = {};

  for (int kt = 0; kt < 25; ++kt) {
    const int k0 = kt * 32;
    // stage K (row-major) and V (transposed); lane owns one key row
    {
      int key = k0 + lane;
      bool ok = key < HWc;
      int keyc = ok ? key : 0;
      const float* krow = kb + (size_t)(b * HWc + keyc) * Dc + h * DHc;
      const float* vrow = vb + (size_t)(b * HWc + keyc) * Dc + h * DHc;
#pragma unroll
      for (int c = 0; c < 32; c += 4) {
        v4f kf = *(const v4f*)&krow[c];
        v4f vf = *(const v4f*)&vrow[c];
        kf = ok ? kf : zero4;
        vf = ok ? vf : zero4;
        *(v4h*)&kl[lane * 32 + c] = __builtin_convertvector(kf, v4h);
#pragma unroll
        for (int j = 0; j < 4; ++j) vlT[(c + j) * 32 + lane] = (_Float16)vf[j];
      }
    }
    __syncthreads();

    // kT B-fragments: key col fixed per lane, dh runs contiguous in kl row
    v16h bk0 = cat8(*(const v8h*)&kl[m * 32 + 8 * g],
                    *(const v8h*)&kl[m * 32 + 16 + 8 * g]);
    v16h bk1 = cat8(*(const v8h*)&kl[(16 + m) * 32 + 8 * g],
                    *(const v8h*)&kl[(16 + m) * 32 + 16 + 8 * g]);
    v8f zc = {};
    v8f s0 = wmma16(aq, bk0, zc);
    v8f s1 = wmma16(aq, bk1, zc);

    float p0[8], p1[8];
#pragma unroll
    for (int r = 0; r < 8; ++r) {
      int qr = q0 + r + 8 * g;
      int key0 = k0 + m, key1 = k0 + 16 + m;
      float x0 = (key0 < HWc) ? (s0[r] * SCALEc + mbase[(size_t)qr * HWc + key0]) : -1e30f;
      float x1 = (key1 < HWc) ? (s1[r] * SCALEc + mbase[(size_t)qr * HWc + key1]) : -1e30f;
      float tmax = fmaxf(x0, x1);
      for (int msk = 1; msk < 16; msk <<= 1)
        tmax = fmaxf(tmax, __shfl_xor(tmax, msk, 32));   // within 16-lane half
      float newm = fmaxf(mrun[r], tmax);
      float corr = __expf(mrun[r] - newm);
      float e0 = __expf(x0 - newm);
      float e1 = __expf(x1 - newm);
      float rs = e0 + e1;
      for (int msk = 1; msk < 16; msk <<= 1)
        rs += __shfl_xor(rs, msk, 32);
      lrun[r] = lrun[r] * corr + rs;
      mrun[r] = newm;
      acc0[r] *= corr;
      acc1[r] *= corr;
      p0[r] = e0;
      p1[r] = e1;
    }

    // repack probs (C layout) -> LDS -> A-fragment layout
#pragma unroll
    for (int r = 0; r < 8; ++r) {
      pl[(r + 8 * g) * 32 + m]      = (_Float16)p0[r];
      pl[(r + 8 * g) * 32 + 16 + m] = (_Float16)p1[r];
    }
    __syncthreads();
    v16h ap = cat8(*(const v8h*)&pl[m * 32 + 8 * g],
                   *(const v8h*)&pl[m * 32 + 16 + 8 * g]);
    // V B-fragments from transposed tile: contiguous key run per dh col
    v16h bv0 = cat8(*(const v8h*)&vlT[m * 32 + 8 * g],
                    *(const v8h*)&vlT[m * 32 + 16 + 8 * g]);
    v16h bv1 = cat8(*(const v8h*)&vlT[(16 + m) * 32 + 8 * g],
                    *(const v8h*)&vlT[(16 + m) * 32 + 16 + 8 * g]);
    acc0 = wmma16(ap, bv0, acc0);
    acc1 = wmma16(ap, bv1, acc1);
    __syncthreads();
  }

#pragma unroll
  for (int r = 0; r < 8; ++r) {
    float inv = (lrun[r] > 0.f) ? (1.f / lrun[r]) : 0.f;
    int row = b * NEc + q0 + r + 8 * g;
    ctx[(size_t)row * Dc + h * DHc + m]      = acc0[r] * inv;
    ctx[(size_t)row * Dc + h * DHc + 16 + m] = acc1[r] * inv;
  }
}

// ---------------------------------------------------------------------------
// Elementwise / reduction helpers
// ---------------------------------------------------------------------------
__global__ void k_ln(const float* __restrict__ x, const float* __restrict__ gg,
                     const float* __restrict__ bb, float* __restrict__ y)
{
  int row = blockIdx.x, t = threadIdx.x;
  __shared__ float red[256];
  float v = x[(size_t)row * Dc + t];
  red[t] = v;
  __syncthreads();
  for (int s = 128; s > 0; s >>= 1) { if (t < s) red[t] += red[t + s]; __syncthreads(); }
  float mean = red[0] * (1.f / Dc);
  __syncthreads();
  float d = v - mean;
  red[t] = d * d;
  __syncthreads();
  for (int s = 128; s > 0; s >>= 1) { if (t < s) red[t] += red[t + s]; __syncthreads(); }
  float var = red[0] * (1.f / Dc);
  y[(size_t)row * Dc + t] = d * rsqrtf(var + 1e-5f) * gg[t] + bb[t];
}

__global__ void k_build_ne(const float* __restrict__ nodes, const float* __restrict__ edges,
                           float* __restrict__ ne)
{
  int i = blockIdx.x * 256 + threadIdx.x;
  if (i >= SR * Dc) return;
  int d = i & 255, s = i >> 8;
  int b = s / NEc, r = s % NEc;
  ne[i] = (r < Nc) ? nodes[((size_t)b * Nc + r) * Dc + d]
                   : edges[((size_t)b * Ec + (r - Nc)) * Dc + d];
}

__global__ void k_slice_nodes(const float* __restrict__ neln, const float* __restrict__ enc,
                              float* __restrict__ out)
{
  int i = blockIdx.x * 256 + threadIdx.x;
  if (i >= BNc * Dc) return;
  int d = i & 255, bn = i >> 8;
  int b = bn >> 5, n = bn & 31;
  out[i] = neln[((size_t)b * NEc + n) * Dc + d] + enc[i];
}

__global__ void k_slice_edges(const float* __restrict__ neln, const float* __restrict__ enc,
                              float* __restrict__ out)
{
  int i = blockIdx.x * 256 + threadIdx.x;
  if (i >= BEc * Dc) return;
  int d = i & 255, be = i >> 8;
  int b = be >> 8, e = be & 255;
  out[i] = neln[((size_t)b * NEc + Nc + e) * Dc + d] + enc[i];
}

__global__ void k_hidden(const float* __restrict__ gn1, const float* __restrict__ gn2,
                         const float* __restrict__ ep, const int* __restrict__ idx0,
                         float* __restrict__ hid)
{
  int i = blockIdx.x * 256 + threadIdx.x;
  if (i >= BEc * Dc) return;
  int d = i & 255, e = i >> 8;
  int n0 = idx0[e];
  hid[i] = geluf(gn1[(size_t)n0 * Dc + d] + gn2[(size_t)n0 * Dc + d] + ep[i]);
}

__global__ void k_pres(const float* __restrict__ hid, const float* __restrict__ ap,
                       float* __restrict__ pres)
{
  int e = blockIdx.x, t = threadIdx.x;
  int h = t >> 5, d = t & 31;
  float v = hid[(size_t)e * Dc + h * DHc + d] * ap[h * DHc + d];
  for (int m = 16; m > 0; m >>= 1) v += __shfl_xor(v, m, 32);
  if (d == 0) pres[e * Hc + h] = v;
}

__global__ void k_fill(float* __restrict__ p, float v, int n)
{
  int i = blockIdx.x * 256 + threadIdx.x;
  if (i < n) p[i] = v;
}

__global__ void k_segmax(const float* __restrict__ pres, const int* __restrict__ idx0,
                         float* __restrict__ mseg)
{
  int i = blockIdx.x * 256 + threadIdx.x;
  if (i >= BEc * Hc) return;
  int e = i >> 3, h = i & 7;
  atomicMaxF(&mseg[idx0[e] * Hc + h], pres[i]);
}

__global__ void k_expv(const float* __restrict__ pres, const int* __restrict__ idx0,
                       const float* __restrict__ mseg, float* __restrict__ alpha,
                       float* __restrict__ den)
{
  int i = blockIdx.x * 256 + threadIdx.x;
  if (i >= BEc * Hc) return;
  int e = i >> 3, h = i & 7;
  int n0 = idx0[e];
  float ev = __expf(pres[i] - mseg[n0 * Hc + h]);
  alpha[i] = ev;
  atomicAdd(&den[n0 * Hc + h], ev);
}

__global__ void k_alpha(float* __restrict__ alpha, const int* __restrict__ idx0,
                        const float* __restrict__ den)
{
  int i = blockIdx.x * 256 + threadIdx.x;
  if (i >= BEc * Hc) return;
  int e = i >> 3, h = i & 7;
  alpha[i] = alpha[i] / den[idx0[e] * Hc + h];
}

__global__ void k_gather_add(const float* __restrict__ nodes_f, const int* __restrict__ idx1,
                             const float* __restrict__ nedges, float* __restrict__ out)
{
  int i = blockIdx.x * 256 + threadIdx.x;
  if (i >= BEc * Dc) return;
  int d = i & 255, e = i >> 8;
  out[i] = nodes_f[(size_t)idx1[e] * Dc + d] + nedges[i];
}

__global__ void k_scatter(const float* __restrict__ vals, const float* __restrict__ alpha,
                          const int* __restrict__ idx0, float* __restrict__ nnodes)
{
  int i = blockIdx.x * 256 + threadIdx.x;
  if (i >= BEc * Dc) return;
  int d = i & 255, e = i >> 8;
  float a = alpha[e * Hc + (d >> 5)];
  atomicAdd(&nnodes[(size_t)idx0[e] * Dc + d], vals[i] * a);
}

__global__ void k_resgat(float* __restrict__ ne, const float* __restrict__ nnodes,
                         const float* __restrict__ nedges, const float* __restrict__ ls2)
{
  int i = blockIdx.x * 256 + threadIdx.x;
  if (i >= SR * Dc) return;
  int d = i & 255, s = i >> 8;
  int b = s / NEc, r = s % NEc;
  float gv = (r < Nc) ? nnodes[((size_t)b * Nc + r) * Dc + d]
                      : nedges[((size_t)b * Ec + (r - Nc)) * Dc + d];
  ne[i] = ne[i] + ls2[d] * gv;
}

// Final copy through LDS using the gfx1250 async global<->LDS path.
__global__ __launch_bounds__(256) void k_copy_async(const float* __restrict__ src,
                                                    float* __restrict__ dst)
{
  __shared__ alignas(16) float buf[1024];
  const int t = threadIdx.x;
  const size_t base = (size_t)blockIdx.x * 1024 + (size_t)t * 4;
  typedef __attribute__((address_space(3))) float lds_f;
  unsigned ldsoff = (unsigned)(uintptr_t)(lds_f*)&buf[t * 4];
  const float* sp = src + base;
  float* dp = dst + base;
  asm volatile("global_load_async_to_lds_b128 %0, %1, off"
               :: "v"(ldsoff), "v"(sp) : "memory");
  asm volatile("s_wait_asynccnt 0" ::: "memory");
  __syncthreads();
  asm volatile("global_store_async_from_lds_b128 %0, %1, off"
               :: "v"(dp), "v"(ldsoff) : "memory");
  asm volatile("s_wait_asynccnt 0" ::: "memory");
}

// ---------------------------------------------------------------------------
static inline void gemm(hipStream_t s, const float* A, int lda, const float* W, int ldw,
                        const float* bias, float* C, int ldc, int M, int Nn, int K,
                        int mode, const float* res, const float* ls)
{
  dim3 g((M + 31) / 32, (Nn + 127) / 128);
  k_gemm<<<g, 128, 0, s>>>(A, lda, W, ldw, bias, C, ldc, M, Nn, K, mode, res, ls);
}

extern "C" void kernel_launch(void* const* d_in, const int* in_sizes, int n_in,
                              void* d_out, int out_size, void* d_ws, size_t ws_size,
                              hipStream_t stream)
{
  auto F = [&](int i) { return (const float*)d_in[i]; };
  const float* nodes    = F(0);
  const float* edges    = F(1);
  const float* images   = F(2);
  const float* mask     = F(3);
  const float* node_enc = F(4);
  const float* edge_enc = F(5);
  const int*   eidx     = (const int*)d_in[34];
  const int*   idx0     = eidx;
  const int*   idx1     = eidx + BEc;

  float* ws = (float*)d_ws;
  size_t off = 0;
  auto alloc = [&](size_t n) { float* p = ws + off; off += n; return p; };
  float* ne      = alloc((size_t)SR * Dc);
  float* neln    = alloc((size_t)SR * Dc);
  float* qb      = alloc((size_t)SR * Dc);
  float* ctx     = alloc((size_t)SR * Dc);
  float* kbv     = alloc((size_t)IR * Dc);   // reused (with vbv) as FFN hidden
  float* vbv     = alloc((size_t)IR * Dc);
  float* nodes_f = alloc((size_t)BNc * Dc);
  float* edges_f = alloc((size_t)BEc * Dc);
  float* gn1     = alloc((size_t)BNc * Dc);
  float* gn2     = alloc((size_t)BNc * Dc);
  float* ep      = alloc((size_t)BEc * Dc);
  float* hid     = alloc((size_t)BEc * Dc);
  float* pres    = alloc((size_t)BEc * Hc);
  float* mseg    = alloc((size_t)BNc * Hc);
  float* den     = alloc((size_t)BNc * Hc);
  float* alpha   = alloc((size_t)BEc * Hc);
  float* nedges  = alloc((size_t)BEc * Dc);
  float* tmpnv   = alloc((size_t)BEc * Dc);
  float* vals    = alloc((size_t)BEc * Dc);
  float* nnodes  = alloc((size_t)BNc * Dc);
  float* h1      = kbv; // FFN hidden [4608,1024] aliases kbv+vbv region

  auto g1 = [](int n) { return dim3((unsigned)((n + 255) / 256)); };

  k_build_ne<<<g1(SR * Dc), 256, 0, stream>>>(nodes, edges, ne);

  for (int l = 0; l < Lc; ++l) {
    const float* ln1g = F(6)  + l * Dc;
    const float* ln1b = F(7)  + l * Dc;
    const float* Wqkv = F(8)  + (size_t)l * Dc * 3 * Dc;
    const float* bqkv = F(9)  + l * 3 * Dc;
    const float* Wo   = F(10) + (size_t)l * Dc * Dc;
    const float* bo   = F(11) + l * Dc;
    const float* ls1  = F(12) + l * Dc;
    const float* ln2g = F(13) + l * Dc;
    const float* ln2b = F(14) + l * Dc;
    const float* Wn1  = F(15) + (size_t)l * Dc * Dc;
    const float* bn1  = F(16) + l * Dc;
    const float* Wn2  = F(17) + (size_t)l * Dc * Dc;
    const float* bn2  = F(18) + l * Dc;
    const float* We   = F(19) + (size_t)l * Dc * Dc;
    const float* be   = F(20) + l * Dc;
    const float* attp = F(21) + l * Hc * DHc;
    const float* Weo  = F(22) + (size_t)l * Dc * Dc;
    const float* beo  = F(23) + l * Dc;
    const float* Wno  = F(24) + (size_t)l * Dc * Dc;
    const float* bno  = F(25) + l * Dc;
    const float* ls2  = F(26) + l * Dc;
    const float* ln3g = F(27) + l * Dc;
    const float* ln3b = F(28) + l * Dc;
    const float* W1   = F(29) + (size_t)l * Dc * FFc;
    const float* b1   = F(30) + l * FFc;
    const float* W2   = F(31) + (size_t)l * FFc * Dc;
    const float* b2   = F(32) + l * Dc;
    const float* ls3  = F(33) + l * Dc;

    // ---- cross-attention ----
    k_ln<<<SR, 256, 0, stream>>>(ne, ln1g, ln1b, neln);
    gemm(stream, neln,   Dc, Wqkv,          3 * Dc, bqkv,          qb,  Dc, SR, Dc, Dc, 0, nullptr, nullptr);
    gemm(stream, images, Dc, Wqkv + Dc,     3 * Dc, bqkv + Dc,     kbv, Dc, IR, Dc, Dc, 0, nullptr, nullptr);
    gemm(stream, images, Dc, Wqkv + 2 * Dc, 3 * Dc, bqkv + 2 * Dc, vbv, Dc, IR, Dc, Dc, 0, nullptr, nullptr);
    k_attn<<<dim3(NEc / 16, Hc, Bc), 32, 0, stream>>>(qb, kbv, vbv, mask, ctx);
    gemm(stream, ctx, Dc, Wo, Dc, bo, ne, Dc, SR, Dc, Dc, 2, ne, ls1);  // ne <- ne1

    // ---- GAT ----
    k_ln<<<SR, 256, 0, stream>>>(ne, ln2g, ln2b, neln);
    k_slice_nodes<<<g1(BNc * Dc), 256, 0, stream>>>(neln, node_enc, nodes_f);
    k_slice_edges<<<g1(BEc * Dc), 256, 0, stream>>>(neln, edge_enc, edges_f);
    gemm(stream, nodes_f, Dc, Wn1, Dc, bn1, gn1, Dc, BNc, Dc, Dc, 0, nullptr, nullptr);
    gemm(stream, nodes_f, Dc, Wn2, Dc, bn2, gn2, Dc, BNc, Dc, Dc, 0, nullptr, nullptr);
    gemm(stream, edges_f, Dc, We,  Dc, be,  ep,  Dc, BEc, Dc, Dc, 0, nullptr, nullptr);
    k_hidden<<<g1(BEc * Dc), 256, 0, stream>>>(gn1, gn2, ep, idx0, hid);
    k_pres<<<BEc, 256, 0, stream>>>(hid, attp, pres);
    k_fill<<<g1(BNc * Hc), 256, 0, stream>>>(mseg, -1e30f, BNc * Hc);
    k_fill<<<g1(BNc * Hc), 256, 0, stream>>>(den, 0.f, BNc * Hc);
    k_segmax<<<g1(BEc * Hc), 256, 0, stream>>>(pres, idx0, mseg);
    k_expv<<<g1(BEc * Hc), 256, 0, stream>>>(pres, idx0, mseg, alpha, den);
    k_alpha<<<g1(BEc * Hc), 256, 0, stream>>>(alpha, idx0, den);
    gemm(stream, hid, Dc, Weo, Dc, beo, nedges, Dc, BEc, Dc, Dc, 0, nullptr, nullptr);
    k_gather_add<<<g1(BEc * Dc), 256, 0, stream>>>(nodes_f, idx1, nedges, tmpnv);
    gemm(stream, tmpnv, Dc, Wno, Dc, bno, vals, Dc, BEc, Dc, Dc, 0, nullptr, nullptr);
    k_fill<<<g1(BNc * Dc), 256, 0, stream>>>(nnodes, 0.f, BNc * Dc);
    k_scatter<<<g1(BEc * Dc), 256, 0, stream>>>(vals, alpha, idx0, nnodes);
    k_resgat<<<g1(SR * Dc), 256, 0, stream>>>(ne, nnodes, nedges, ls2);  // ne <- ne2

    // ---- FFN ----
    k_ln<<<SR, 256, 0, stream>>>(ne, ln3g, ln3b, neln);
    gemm(stream, neln, Dc, W1, FFc, b1, h1, FFc, SR, FFc, Dc, 1, nullptr, nullptr);
    gemm(stream, h1, FFc, W2, Dc, b2, ne, Dc, SR, Dc, FFc, 2, ne, ls3);  // ne <- ne3
  }

  // SR*Dc = 1179648 floats = 1152 blocks * 1024 floats
  k_copy_async<<<dim3((SR * Dc) / 1024), 256, 0, stream>>>(ne, (float*)d_out);
}
